// UDLSLinear_76063870812659
// MI455X (gfx1250) — matrix operand order repaired and checked by reading, not compiled, not run on hardware
//
#include <hip/hip_runtime.h>
#include <hip/hip_bf16.h>

// ---------------- problem sizes (match reference) ----------------
#define IN_F   4096
#define OUT_F  4096
#define GROUP  8
#define NUM_GROUPS (IN_F * OUT_F / GROUP)   // 2,097,152
#define M_TOT  8192                          // 4 * 2048 rows of x

// ---------------- GEMM tiling ----------------
#define BM   128
#define BN   128
#define BK   32            // one v_wmma_f32_16x16x32_bf16 K-step
#define BKP  40            // padded row (halves): 80 B stride, 16B-aligned
#define NKT  (IN_F / BK)
#define BUF_BYTES (BM * BKP * 2)   // 10240 B per stage per array

typedef __attribute__((ext_vector_type(8)))  __bf16 v8bf;
typedef __attribute__((ext_vector_type(16))) __bf16 v16bf;
typedef __attribute__((ext_vector_type(8)))  float  v8f;

// ----- f32 -> bf16 round-to-nearest-even (off the hot path) -----
__device__ __forceinline__ unsigned short f32_to_bf16(float f) {
    union { float f; unsigned u; } v; v.f = f;
    unsigned r = v.u + 0x7FFFu + ((v.u >> 16) & 1u);
    return (unsigned short)(r >> 16);
}
__device__ __forceinline__ unsigned pack_bf2(float a, float b) {
    return (unsigned)f32_to_bf16(a) | ((unsigned)f32_to_bf16(b) << 16);
}
// cheap pack for the fallback: +0x7FFF nearest + one v_perm_b32
__device__ __forceinline__ unsigned pack_rn(float a, float b) {
    unsigned ua = __float_as_uint(a) + 0x7FFFu;
    unsigned ub = __float_as_uint(b) + 0x7FFFu;
    return __builtin_amdgcn_perm(ub, ua, 0x07060302u);
}

// ----- CDNA5 async global->LDS copy (ASYNCcnt-tracked, no VGPR staging) -----
__device__ __forceinline__ void async_cp_b128(unsigned lds_off, const void* gptr) {
    asm volatile("global_load_async_to_lds_b128 %0, %1, off"
                 :: "v"(lds_off), "v"((unsigned long long)(size_t)gptr)
                 : "memory");
}
__device__ __forceinline__ void wait_async0() {
    asm volatile("s_wait_asynccnt 0x0" ::: "memory");
}

// =====================================================================
// Kernel 1: decode quantized weight -> bf16 W[OUT_F][IN_F] in workspace
// =====================================================================
__global__ __launch_bounds__(256)
void decode_kernel(const float* __restrict__ codebook,
                   const int*   __restrict__ C,
                   const float* __restrict__ S,
                   const float* __restrict__ R,
                   unsigned short* __restrict__ Wbf) {
    const int g = blockIdx.x * 256 + threadIdx.x;   // grid sized exactly
    const int idx = C[g];
    const float s = S[g];

    const float4* cb = (const float4*)codebook + (size_t)idx * 2;
    const float4  c0 = cb[0], c1 = cb[1];
    const float4* rp = (const float4*)R + (size_t)g * 2;
    const float4  r0 = rp[0], r1 = rp[1];

    uint4 o;
    o.x = pack_bf2(c0.x * s + r0.x, c0.y * s + r0.y);
    o.y = pack_bf2(c0.z * s + r0.z, c0.w * s + r0.w);
    o.z = pack_bf2(c1.x * s + r1.x, c1.y * s + r1.y);
    o.w = pack_bf2(c1.z * s + r1.z, c1.w * s + r1.w);
    ((uint4*)Wbf)[g] = o;
}

// =====================================================================
// Kernel 2: one-time x (f32) -> bf16 pre-pass (memory-bound)
// =====================================================================
__global__ __launch_bounds__(256)
void cvt_x_kernel(const float* __restrict__ x, unsigned short* __restrict__ xbf) {
    const size_t g = (size_t)blockIdx.x * 256 + threadIdx.x;  // one per 8 elems
    const float4* p = (const float4*)x + g * 2;
    const float4 a = p[0], b = p[1];
    uint4 o;
    o.x = pack_bf2(a.x, a.y);
    o.y = pack_bf2(a.z, a.w);
    o.z = pack_bf2(b.x, b.y);
    o.w = pack_bf2(b.z, b.w);
    ((uint4*)xbf)[g] = o;
}

// =====================================================================
// Kernel 3 (main): y = x @ W^T + bias, bf16 WMMA, async global->LDS staging
// Block: 256 threads = 8 waves (2x4), tile 128x128, wave tile 64x32.
// Double-buffered LDS; 4 async b128 copies per thread per K-tile;
// one s_wait_asynccnt + one barrier per K-tile.
// =====================================================================
__global__ __launch_bounds__(256)
void gemm_async_kernel(const unsigned short* __restrict__ X,
                       const unsigned short* __restrict__ W,
                       const float*          __restrict__ bias,
                       float*                __restrict__ out) {
    __shared__ __align__(16) unsigned short As[2][BM][BKP];
    __shared__ __align__(16) unsigned short Bs[2][BN][BKP];

    const int tid  = threadIdx.x;
    const int lane = tid & 31;
    const int wave = tid >> 5;
    const int wm   = wave >> 2;            // 0..1  -> 64 rows each
    const int wn   = wave & 3;             // 0..3  -> 32 cols each
    const int l15  = lane & 15;
    const int hi   = lane >> 4;

    const int m0 = blockIdx.y * BM;
    const int n0 = blockIdx.x * BN;
    const int wmBase = wm * 64;
    const int wnBase = wn * 32;

    v8f acc[4][2];
#pragma unroll
    for (int i = 0; i < 4; ++i)
#pragma unroll
        for (int j = 0; j < 2; ++j) acc[i][j] = (v8f)0.0f;

    float bias_n[2];
    bias_n[0] = bias[n0 + wnBase + l15];
    bias_n[1] = bias[n0 + wnBase + 16 + l15];

    // ---- staging geometry: 2 threads per row, 16 contiguous halves each ----
    const int rowL = tid >> 1;             // 0..127
    const int kseg = (tid & 1) * 16;       // 0 or 16 (halves)
    const unsigned short* gA = X + (size_t)(m0 + rowL) * IN_F + kseg;
    const unsigned short* gB = W + (size_t)(n0 + rowL) * IN_F + kseg;

    // LDS byte offsets (low 32 bits of a generic pointer to __shared__ = LDS offset)
    const unsigned ldsA0 = (unsigned)(size_t)&As[0][rowL][kseg];
    const unsigned ldsB0 = (unsigned)(size_t)&Bs[0][rowL][kseg];

    auto issue_tile = [&](int kt, int buf) {
        const unsigned short* pa = gA + (size_t)kt * BK;
        const unsigned short* pb = gB + (size_t)kt * BK;
        const unsigned la = ldsA0 + (unsigned)buf * BUF_BYTES;
        const unsigned lb = ldsB0 + (unsigned)buf * BUF_BYTES;
        async_cp_b128(la,      pa);
        async_cp_b128(la + 16, pa + 8);
        async_cp_b128(lb,      pb);
        async_cp_b128(lb + 16, pb + 8);
    };

    issue_tile(0, 0);
    wait_async0();
    __syncthreads();

    for (int kt = 0; kt < NKT; ++kt) {
        const int cur = kt & 1;
        const bool more = (kt + 1) < NKT;

        // async copies for the next K-tile run while WMMAs execute
        if (more) issue_tile(kt + 1, cur ^ 1);

        // ---- B fragments: lane l15 = column n; lanes 0-15: K 0..15, 16-31: K 16..31 ----
        v16bf bfrag[2];
#pragma unroll
        for (int j = 0; j < 2; ++j) {
            const unsigned short* pb = &Bs[cur][wnBase + j * 16 + l15][hi ? 16 : 0];
            union { v16bf v; v8bf h[2]; } u;
            u.h[0] = *(const v8bf*)(pb);
            u.h[1] = *(const v8bf*)(pb + 8);
            bfrag[j] = u.v;
        }

        // ---- A fragments + WMMA ----
#pragma unroll
        for (int i = 0; i < 4; ++i) {
            const unsigned short* pa = &As[cur][wmBase + i * 16 + l15][hi ? 16 : 0];
            union { v16bf v; v8bf h[2]; } ua;
            ua.h[0] = *(const v8bf*)(pa);
            ua.h[1] = *(const v8bf*)(pa + 8);
            const v16bf afrag = ua.v;
#pragma unroll
            for (int j = 0; j < 2; ++j) {
                acc[i][j] = __builtin_amdgcn_wmma_f32_16x16x32_bf16(
                    false, afrag, false, bfrag[j],
                    (short)0, acc[i][j], false, false);
            }
        }

        if (more) wait_async0();   // my async writes to LDS[cur^1] landed
        __syncthreads();           // everyone's landed; safe to read next buffer
    }

    // ---- epilogue: C/D layout is VGPR r -> (M = r + 8*hi, N = l15) ----
#pragma unroll
    for (int i = 0; i < 4; ++i) {
#pragma unroll
        for (int j = 0; j < 2; ++j) {
            const int mrow = m0 + wmBase + i * 16 + (hi ? 8 : 0);
            const int ncol = n0 + wnBase + j * 16 + l15;
            const float bn = bias_n[j];
#pragma unroll
            for (int r = 0; r < 8; ++r) {
                out[(size_t)(mrow + r) * OUT_F + ncol] = acc[i][j][r] + bn;
            }
        }
    }
}

// =====================================================================
// Kernel 3 (fallback, ws too small): f32 A converted in-loop, single buffer
// =====================================================================
__global__ __launch_bounds__(256)
void gemm_f32_kernel(const float*          __restrict__ x,
                     const unsigned short* __restrict__ W,
                     const float*          __restrict__ bias,
                     float*                __restrict__ out) {
    __shared__ __align__(16) unsigned short As[BM][BKP];
    __shared__ __align__(16) unsigned short Bs[BN][BKP];

    const int tid  = threadIdx.x;
    const int lane = tid & 31;
    const int wave = tid >> 5;
    const int wm   = wave >> 2;
    const int wn   = wave & 3;
    const int l15  = lane & 15;
    const int hi   = lane >> 4;

    const int m0 = blockIdx.y * BM;
    const int n0 = blockIdx.x * BN;
    const int wmBase = wm * 64;
    const int wnBase = wn * 32;

    v8f acc[4][2];
#pragma unroll
    for (int i = 0; i < 4; ++i)
#pragma unroll
        for (int j = 0; j < 2; ++j) acc[i][j] = (v8f)0.0f;

    float bias_n[2];
    bias_n[0] = bias[n0 + wnBase + l15];
    bias_n[1] = bias[n0 + wnBase + 16 + l15];

    const int rowL = tid >> 1;
    const int kseg = (tid & 1) * 16;
    const float*          gA = x + (size_t)(m0 + rowL) * IN_F + kseg;
    const unsigned short* gB = W + (size_t)(n0 + rowL) * IN_F + kseg;

    for (int kt = 0; kt < NKT; ++kt) {
        const float* pA = gA + (size_t)kt * BK;
        float4 aF[4];
#pragma unroll
        for (int i = 0; i < 4; ++i) aF[i] = *(const float4*)(pA + 4 * i);
        const unsigned short* pB = gB + (size_t)kt * BK;
        const uint4 b0 = *(const uint4*)(pB);
        const uint4 b1 = *(const uint4*)(pB + 8);

        __syncthreads();   // previous compute done before overwriting
        uint4 u0, u1;
        u0.x = pack_rn(aF[0].x, aF[0].y); u0.y = pack_rn(aF[0].z, aF[0].w);
        u0.z = pack_rn(aF[1].x, aF[1].y); u0.w = pack_rn(aF[1].z, aF[1].w);
        u1.x = pack_rn(aF[2].x, aF[2].y); u1.y = pack_rn(aF[2].z, aF[2].w);
        u1.z = pack_rn(aF[3].x, aF[3].y); u1.w = pack_rn(aF[3].z, aF[3].w);
        *(uint4*)&As[rowL][kseg]     = u0;
        *(uint4*)&As[rowL][kseg + 8] = u1;
        *(uint4*)&Bs[rowL][kseg]     = b0;
        *(uint4*)&Bs[rowL][kseg + 8] = b1;
        __syncthreads();

        v16bf bfrag[2];
#pragma unroll
        for (int j = 0; j < 2; ++j) {
            const unsigned short* pb = &Bs[wnBase + j * 16 + l15][hi ? 16 : 0];
            union { v16bf v; v8bf h[2]; } u;
            u.h[0] = *(const v8bf*)(pb);
            u.h[1] = *(const v8bf*)(pb + 8);
            bfrag[j] = u.v;
        }
#pragma unroll
        for (int i = 0; i < 4; ++i) {
            const unsigned short* pa = &As[wmBase + i * 16 + l15][hi ? 16 : 0];
            union { v16bf v; v8bf h[2]; } ua;
            ua.h[0] = *(const v8bf*)(pa);
            ua.h[1] = *(const v8bf*)(pa + 8);
            const v16bf afrag = ua.v;
#pragma unroll
            for (int j = 0; j < 2; ++j) {
                acc[i][j] = __builtin_amdgcn_wmma_f32_16x16x32_bf16(
                    false, afrag, false, bfrag[j],
                    (short)0, acc[i][j], false, false);
            }
        }
    }

#pragma unroll
    for (int i = 0; i < 4; ++i) {
#pragma unroll
        for (int j = 0; j < 2; ++j) {
            const int mrow = m0 + wmBase + i * 16 + (hi ? 8 : 0);
            const int ncol = n0 + wnBase + j * 16 + l15;
            const float bn = bias_n[j];
#pragma unroll
            for (int r = 0; r < 8; ++r) {
                out[(size_t)(mrow + r) * OUT_F + ncol] = acc[i][j][r] + bn;
            }
        }
    }
}

// =====================================================================
extern "C" void kernel_launch(void* const* d_in, const int* in_sizes, int n_in,
                              void* d_out, int out_size, void* d_ws, size_t ws_size,
                              hipStream_t stream) {
    const float* x        = (const float*)d_in[0];  // [4,2048,4096]
    const float* codebook = (const float*)d_in[1];  // [256,8]
    const int*   C        = (const int*)d_in[2];    // [NUM_GROUPS]
    const float* S        = (const float*)d_in[3];  // [NUM_GROUPS]
    const float* R        = (const float*)d_in[4];  // [OUT_F,IN_F]
    const float* bias     = (const float*)d_in[5];  // [OUT_F]
    float*       out      = (float*)d_out;          // [4,2048,4096]

    const size_t wBytes = (size_t)OUT_F * IN_F * sizeof(unsigned short);  // 32 MB
    const size_t xBytes = (size_t)M_TOT * IN_F * sizeof(unsigned short);  // 64 MB

    unsigned short* Wbf = (unsigned short*)d_ws;
    decode_kernel<<<NUM_GROUPS / 256, 256, 0, stream>>>(codebook, C, S, R, Wbf);

    dim3 grid(OUT_F / BN, M_TOT / BM);              // (32, 64)

    if (ws_size >= wBytes + xBytes) {
        unsigned short* Xbf = (unsigned short*)((char*)d_ws + wBytes);
        cvt_x_kernel<<<(int)((size_t)M_TOT * IN_F / 8 / 256), 256, 0, stream>>>(x, Xbf);
        gemm_async_kernel<<<grid, 256, 0, stream>>>(Xbf, Wbf, bias, out);
    } else {
        gemm_f32_kernel<<<grid, 256, 0, stream>>>(x, Wbf, bias, out);
    }
}
